// GAT_78632261256143
// MI455X (gfx1250) — matrix operand order repaired and checked by reading, not compiled, not run on hardware
//
#include <hip/hip_runtime.h>

// ---- CDNA5 / gfx1250 vector types for WMMA --------------------------------
typedef __attribute__((ext_vector_type(16))) __bf16 v16bf;
typedef __attribute__((ext_vector_type(8)))  __bf16 v8bf;
typedef __attribute__((ext_vector_type(8)))  float  v8f;
typedef __attribute__((ext_vector_type(8)))  int    v8i;

static constexpr int BB    = 4;
static constexpr int NN    = 2048;
static constexpr int CIN   = 64;
static constexpr int HID   = 128;
static constexpr int HEADS = 4;
static constexpr int DD    = 32;     // per-head dim
static constexpr int NQ    = 16;
static constexpr float NEG_SLOPE = 0.2f;
static constexpr float MASK_VAL  = -1.0e9f;

// ---------------------------------------------------------------------------
// W_in [64][128] f32  ->  WT [128][64] bf16 (column-major for B-operand loads)
__global__ void k_prep_wt(const float* __restrict__ W_in, __bf16* __restrict__ WT) {
    int idx = blockIdx.x * blockDim.x + threadIdx.x;
    if (idx < CIN * HID) {
        int n = idx >> 6;        // 0..127
        int k = idx & 63;        // 0..63
        WT[idx] = (__bf16)W_in[k * HID + n];
    }
}

// ---------------------------------------------------------------------------
// h = x @ W_in + b_in   via v_wmma_f32_16x16x32_bf16
// one wave per 16x16 output tile; M = B*N = 8192, K = 64, N = 128
__global__ void __launch_bounds__(256) k_gemm_in(
        const float* __restrict__ x, const float* __restrict__ b_in,
        const __bf16* __restrict__ WT,
        float* __restrict__ h, __bf16* __restrict__ hT) {
    int wave = (blockIdx.x * blockDim.x + threadIdx.x) >> 5;
    int lane = threadIdx.x & 31;
    int hi   = lane >> 4;          // half-wave selector (K interleave)
    int lr   = lane & 15;
    int nt   = wave & 7;           // 8 column tiles (HID/16)
    int mt   = wave >> 3;          // 512 row tiles  (B*N/16)
    int mrow = mt * 16 + lr;       // A-operand row for this lane
    int col  = nt * 16 + lr;       // B-operand column / C column for this lane

    const float*  xrow = x  + (size_t)mrow * CIN;
    const __bf16* wcol = WT + (size_t)col  * CIN;

    v8f acc = {};
    #pragma unroll
    for (int kb = 0; kb < CIN; kb += 32) {
        // A 16x32 bf16 layout: lane<16 holds K = kb+{0..7, 16..23}; lane>=16 +8
        v8f a0 = *(const v8f*)(xrow + kb + hi * 8);
        v8f a1 = *(const v8f*)(xrow + kb + 16 + hi * 8);
        v16bf A;
        #pragma unroll
        for (int t = 0; t < 8; ++t) { A[t] = (__bf16)a0[t]; A[t + 8] = (__bf16)a1[t]; }
        // B 32x16 bf16 layout: lane<16 holds K = kb+{0..7, 16..23}; lane>=16 +8
        v16bf Bv = __builtin_shufflevector(*(const v8bf*)(wcol + kb + hi * 8), *(const v8bf*)(wcol + kb + 16 + hi * 8), 0,1,2,3,4,5,6,7,8,9,10,11,12,13,14,15);
        acc = __builtin_amdgcn_wmma_f32_16x16x32_bf16(false, A, false, Bv,
                                                      (short)0, acc, false, false);
    }

    float bias = b_in[col];
    int head = col >> 5, d = col & 31;
    #pragma unroll
    for (int g = 0; g < 8; ++g) {
        int row = mt * 16 + g + hi * 8;          // C layout: VGPR g -> row g / g+8
        int b   = row >> 11;
        float v = acc[g] + bias;
        h[(size_t)row * HID + col] = v;
        // transposed bf16 values for attention B-operand: hT[b][head][d][j]
        hT[(((size_t)b * HEADS + head) * DD + d) * NN + (row & (NN - 1))] = (__bf16)v;
    }
}

// ---------------------------------------------------------------------------
// alpha_src/dst[b,h,n] = sum_d h[b,n,h,d] * att_{src,dst}[h,d]   (layout [B,H,N])
__global__ void k_alpha(const float* __restrict__ h,
                        const float* __restrict__ att_src,
                        const float* __restrict__ att_dst,
                        float* __restrict__ as_, float* __restrict__ ad_) {
    int idx = blockIdx.x * blockDim.x + threadIdx.x;   // b*N + n
    if (idx >= BB * NN) return;
    const float* hr = h + (size_t)idx * HID;
    int b = idx >> 11, n = idx & (NN - 1);
    #pragma unroll
    for (int hd = 0; hd < HEADS; ++hd) {
        float s = 0.f, t = 0.f;
        #pragma unroll
        for (int d = 0; d < DD; ++d) {
            float v = hr[hd * DD + d];
            s += v * att_src[hd * DD + d];
            t += v * att_dst[hd * DD + d];
        }
        as_[((size_t)b * HEADS + hd) * NN + n] = s;
        ad_[((size_t)b * HEADS + hd) * NN + n] = t;
    }
}

// ---------------------------------------------------------------------------
// Fused masked-softmax attention aggregation (flash-attention style).
// One wave per (b, head, 16-row i-tile); streams j in blocks of 32;
// P tile built directly in WMMA A-layout; 2 WMMAs per block (D=32).
__global__ void __launch_bounds__(128) k_attn(
        const __bf16* __restrict__ hT, const float* __restrict__ as_,
        const float* __restrict__ ad_, const int* __restrict__ adj,
        float* __restrict__ out_attn) {
    int wave = (blockIdx.x * blockDim.x + threadIdx.x) >> 5;
    int lane = threadIdx.x & 31;
    int hi   = lane >> 4;
    int lr   = lane & 15;
    int it   = wave & 127;            // i tile
    int head = (wave >> 7) & 3;
    int b    = wave >> 9;

    int i = it * 16 + lr;             // this lane's score row
    float asv = as_[((size_t)b * HEADS + head) * NN + i];
    const float*  adp    = ad_ + ((size_t)b * HEADS + head) * NN;
    const int*    adjrow = adj + (size_t)i * NN;
    const __bf16* hTb    = hT + ((size_t)b * HEADS + head) * DD * NN;
    int d0 = lr, d1 = lr + 16;        // C-operand columns for the two accumulators

    v8f acc0 = {}, acc1 = {};
    float m = -__builtin_inff(), l = 0.f;
    int off = hi * 8;

    for (int jb = 0; jb < NN; jb += 32) {
        // score inputs for this lane's K subset (A-layout interleave)
        v8f ad0 = *(const v8f*)(adp + jb + off);
        v8f ad1 = *(const v8f*)(adp + jb + 16 + off);
        v8i aj0 = *(const v8i*)(adjrow + jb + off);
        v8i aj1 = *(const v8i*)(adjrow + jb + 16 + off);

        float s[16];
        #pragma unroll
        for (int t = 0; t < 8; ++t) {
            float v0 = asv + ad0[t];
            v0 = v0 >= 0.f ? v0 : NEG_SLOPE * v0;
            s[t] = (aj0[t] != 0) ? v0 : MASK_VAL;
            float v1 = asv + ad1[t];
            v1 = v1 >= 0.f ? v1 : NEG_SLOPE * v1;
            s[t + 8] = (aj1[t] != 0) ? v1 : MASK_VAL;
        }

        // online softmax: row stats live (duplicated) in lane pairs (lr, lr+16)
        float bm = s[0];
        #pragma unroll
        for (int t = 1; t < 16; ++t) bm = fmaxf(bm, s[t]);
        bm = fmaxf(bm, __shfl_xor(bm, 16));
        float mn    = fmaxf(m, bm);
        float scale = __expf(m - mn);
        m = mn;

        float ps = 0.f;
        v16bf A;
        #pragma unroll
        for (int t = 0; t < 16; ++t) {
            float p = __expf(s[t] - mn);
            ps  += p;
            A[t] = (__bf16)p;
        }
        ps += __shfl_xor(ps, 16);
        l = l * scale + ps;

        // rescale accumulators: VGPR g holds row g (lanes<16) / g+8 (lanes>=16)
        #pragma unroll
        for (int g = 0; g < 8; ++g) {
            float f = __shfl(scale, g + off);
            acc0[g] *= f;
            acc1[g] *= f;
        }

        // B-operand: j = jb+{0..7, 16..23} (+8 for lanes>=16) from transposed hT
        v16bf B0 = __builtin_shufflevector(*(const v8bf*)(hTb + (size_t)d0 * NN + jb + off), *(const v8bf*)(hTb + (size_t)d0 * NN + jb + 16 + off), 0,1,2,3,4,5,6,7,8,9,10,11,12,13,14,15);
        v16bf B1 = __builtin_shufflevector(*(const v8bf*)(hTb + (size_t)d1 * NN + jb + off), *(const v8bf*)(hTb + (size_t)d1 * NN + jb + 16 + off), 0,1,2,3,4,5,6,7,8,9,10,11,12,13,14,15);
        acc0 = __builtin_amdgcn_wmma_f32_16x16x32_bf16(false, A, false, B0,
                                                       (short)0, acc0, false, false);
        acc1 = __builtin_amdgcn_wmma_f32_16x16x32_bf16(false, A, false, B1,
                                                       (short)0, acc1, false, false);
    }

    float linv = 1.0f / l;
    #pragma unroll
    for (int g = 0; g < 8; ++g) {
        float f  = __shfl(linv, g + off);
        int row  = it * 16 + g + hi * 8;
        size_t base = (((size_t)b * NN + row) * HEADS + head) * DD;
        out_attn[base + d0] = acc0[g] * f;
        out_attn[base + d1] = acc1[g] * f;
    }
}

// ---------------------------------------------------------------------------
__global__ void k_out_init(const float* __restrict__ b_out, float* __restrict__ out) {
    int t = threadIdx.x;
    if (t < BB * NQ) out[t] = b_out[t & 15];
}

// out[4][16] += out_attn[4][262144] @ W_out[262144][16]
__global__ void __launch_bounds__(256) k_gemm_out(
        const float* __restrict__ oa, const float* __restrict__ W_out,
        float* __restrict__ out) {
    __shared__ float sdata[256];
    const int K = NN * HID;
    int t  = threadIdx.x;
    int q  = t & 15;
    int kk = t >> 4;                       // 0..15
    int kchunk = K / gridDim.x;            // grid 128 -> 2048 k's per block
    int kbase  = blockIdx.x * kchunk;
    float acc[BB] = {0.f, 0.f, 0.f, 0.f};
    for (int k = kbase + kk; k < kbase + kchunk; k += 16) {
        float w = W_out[(size_t)k * NQ + q];   // coalesced across the block
        #pragma unroll
        for (int mi = 0; mi < BB; ++mi)
            acc[mi] += oa[(size_t)mi * K + k] * w;
    }
    #pragma unroll
    for (int mi = 0; mi < BB; ++mi) {
        sdata[t] = acc[mi];
        __syncthreads();
        for (int s2 = 128; s2 >= 16; s2 >>= 1) {
            if (t < s2) sdata[t] += sdata[t + s2];
            __syncthreads();
        }
        if (t < 16) atomicAdd(&out[mi * NQ + t], sdata[t]);
        __syncthreads();
    }
}

// ---------------------------------------------------------------------------
extern "C" void kernel_launch(void* const* d_in, const int* in_sizes, int n_in,
                              void* d_out, int out_size, void* d_ws, size_t ws_size,
                              hipStream_t stream) {
    (void)in_sizes; (void)n_in; (void)out_size; (void)ws_size;
    const float* x       = (const float*)d_in[0];
    const int*   adj     = (const int*)  d_in[1];
    const float* W_in    = (const float*)d_in[2];
    const float* b_in    = (const float*)d_in[3];
    const float* att_src = (const float*)d_in[4];
    const float* att_dst = (const float*)d_in[5];
    const float* W_out   = (const float*)d_in[6];
    const float* b_out   = (const float*)d_in[7];
    float* out = (float*)d_out;

    // workspace partition (all offsets 256B aligned)
    char* ws = (char*)d_ws;
    float*  h   = (float*) (ws);                               // 4 MB
    __bf16* hT  = (__bf16*)(ws + (size_t)4  * 1024 * 1024);    // 2 MB
    float*  as_ = (float*) (ws + (size_t)6  * 1024 * 1024);    // 128 KB
    float*  ad_ = (float*) (ws + (size_t)6  * 1024 * 1024 + 128 * 1024);
    __bf16* WT  = (__bf16*)(ws + (size_t)6  * 1024 * 1024 + 256 * 1024);  // 16 KB
    float*  oa  = (float*) (ws + (size_t)6  * 1024 * 1024 + 512 * 1024);  // 4 MB

    k_prep_wt <<<32,  256, 0, stream>>>(W_in, WT);
    k_gemm_in <<<512, 256, 0, stream>>>(x, b_in, WT, h, hT);
    k_alpha   <<<32,  256, 0, stream>>>(h, att_src, att_dst, as_, ad_);
    k_attn    <<<512, 128, 0, stream>>>(hT, as_, ad_, adj, oa);
    k_out_init<<<1,   64,  0, stream>>>(b_out, out);
    k_gemm_out<<<128, 256, 0, stream>>>(oa, W_out, out);
}
